// EdgeGAT_54597624267050
// MI455X (gfx1250) — compile-verified, matrix operand order checked by reading, not run on hardware
//
#include <hip/hip_runtime.h>
#include <hip/hip_bf16.h>

typedef __attribute__((ext_vector_type(16))) __bf16 v16bf;
typedef __attribute__((ext_vector_type(8)))  __bf16 v8bf;
typedef __attribute__((ext_vector_type(8)))  float  v8f;

#define NSLOPE 0.2f
#define BN_EPS 1e-5f
#define HD 256
#define HID 64
#define NHEAD 4

__device__ __forceinline__ float lrelu(float v) { return v > 0.f ? v : NSLOPE * v; }

// order-preserving float<->uint encoding for atomicMax on floats
__device__ __forceinline__ unsigned fflip(float f) {
  unsigned u = __float_as_uint(f);
  return (u & 0x80000000u) ? ~u : (u | 0x80000000u);
}
__device__ __forceinline__ float funflip(unsigned u) {
  return __uint_as_float((u & 0x80000000u) ? (u & 0x7fffffffu) : ~u);
}

// Load one 16-bit WMMA operand fragment for this lane.
// Per CDNA5 ISA 16-bit A layout: lane half h = lane>>4 selects K groups
// [8h .. 8h+7] (elements 0..7) and [16+8h .. 16+8h+7] (elements 8..15),
// both contiguous in K -> two 16-byte loads. Caller passes p = row + kb + 8*h.
__device__ __forceinline__ v16bf ld_frag(const __bf16* p) {
  v8bf lo = *(const v8bf*)(p);
  v8bf hi = *(const v8bf*)(p + 16);
  v16bf r;
#pragma unroll
  for (int i = 0; i < 8; ++i) { r[i] = lo[i]; r[i + 8] = hi[i]; }
  return r;
}

// ---------------- weight packing / casts ----------------

// W [K][C] row-major fp32  ->  Wt [C][Kpad] bf16 (zero padded K)
__global__ void k_pack_wt(const float* __restrict__ W, __bf16* __restrict__ Wt,
                          int K, int C, int Kpad) {
  int i = blockIdx.x * blockDim.x + threadIdx.x;
  if (i >= C * Kpad) return;
  int c = i / Kpad, k = i - c * Kpad;
  Wt[i] = (__bf16)(k < K ? W[(size_t)k * C + c] : 0.0f);
}

__global__ void k_cast_bf16(const float* __restrict__ X, __bf16* __restrict__ Y, int n) {
  int i = blockIdx.x * blockDim.x + threadIdx.x;
  if (i < n) Y[i] = (__bf16)X[i];
}

__global__ void k_zero(float* p, int n) {
  int i = blockIdx.x * blockDim.x + threadIdx.x;
  if (i < n) p[i] = 0.f;
}

// ---------------- WMMA GEMM: C[M][Ncols] = A[M][K](bf16) x Bt[Ncols][K](bf16) -------------
// Each wave computes a 16x64 strip (4 accumulators) to reuse the A fragment
// 4x per K-step: operand traffic drops from 64B to 40B per lane per wmma.
template <int K>
__global__ __launch_bounds__(256) void k_gemm_bf16(const __bf16* __restrict__ A,
                                                   const __bf16* __restrict__ Bt,
                                                   float* __restrict__ C,
                                                   int M, int Ncols) {
  int wave = blockIdx.x * (blockDim.x >> 5) + (threadIdx.x >> 5);
  int ntn = Ncols >> 6;                 // 64-wide column strips
  int tiles = (M >> 4) * ntn;
  if (wave >= tiles) return;            // wave-uniform: EXEC stays all-ones for WMMA
  int tm = wave / ntn, tn = wave - tm * ntn;
  int lane = threadIdx.x & 31;
  int hh = lane >> 4, ln = lane & 15;
  const __bf16* arow = A + (size_t)(tm * 16 + ln) * K + 8 * hh;
  const __bf16* brow = Bt + (size_t)(tn * 64 + ln) * K + 8 * hh;
  v8f acc[4] = {v8f{}, v8f{}, v8f{}, v8f{}};
#pragma unroll
  for (int kb = 0; kb < K; kb += 32) {
    v16bf a = ld_frag(arow + kb);
#pragma unroll
    for (int j = 0; j < 4; ++j) {
      v16bf b = ld_frag(brow + (size_t)(j * 16) * K + kb);
      acc[j] = __builtin_amdgcn_wmma_f32_16x16x32_bf16(false, a, false, b, (short)0,
                                                       acc[j], false, false);
    }
  }
  float* crow = C + (size_t)(tm * 16 + 8 * hh) * Ncols + tn * 64 + ln;
#pragma unroll
  for (int j = 0; j < 4; ++j)
#pragma unroll
    for (int r = 0; r < 8; ++r) crow[(size_t)r * Ncols + j * 16] = acc[j][r];
}

// ---------------- GAT attention kernels ----------------

// alpha_src/dst per (node,head); init segmax with self-loop score (flipped uint)
__global__ void k_alpha(const float* __restrict__ h, const float* __restrict__ a_s,
                        const float* __restrict__ a_d, float* __restrict__ asrc,
                        float* __restrict__ adst, unsigned* __restrict__ segmax, int n) {
  int i = blockIdx.x * blockDim.x + threadIdx.x;
  if (i >= n * NHEAD) return;
  int node = i >> 2, hh = i & 3;
  const float* hp = h + (size_t)node * HD + hh * HID;
  float s = 0.f, d = 0.f;
#pragma unroll 8
  for (int k = 0; k < HID; ++k) {
    float v = hp[k];
    s += v * a_s[hh * HID + k];
    d += v * a_d[hh * HID + k];
  }
  asrc[i] = s;
  adst[i] = d;
  segmax[i] = fflip(lrelu(s + d));
}

__global__ void k_edge_max(const int* __restrict__ ei, const float* __restrict__ asrc,
                           const float* __restrict__ adst, unsigned* __restrict__ segmax,
                           int nE) {
  int i = blockIdx.x * blockDim.x + threadIdx.x;
  if (i >= nE * NHEAD) return;
  int e = i >> 2, hh = i & 3;
  int s = ei[e], d = ei[nE + e];
  float v = lrelu(asrc[s * NHEAD + hh] + adst[d * NHEAD + hh]);
  atomicMax(&segmax[d * NHEAD + hh], fflip(v));
}

// convert segmax in-place (uint->float) and seed segsum with self-loop term
__global__ void k_node_expinit(const float* __restrict__ asrc, const float* __restrict__ adst,
                               unsigned* __restrict__ segmax, float* __restrict__ segsum,
                               int n) {
  int i = blockIdx.x * blockDim.x + threadIdx.x;
  if (i >= n * NHEAD) return;
  float m = funflip(segmax[i]);
  ((float*)segmax)[i] = m;
  float es = lrelu(asrc[i] + adst[i]);
  segsum[i] = __expf(es - m);
}

__global__ void k_edge_sum(const int* __restrict__ ei, const float* __restrict__ asrc,
                           const float* __restrict__ adst, const float* __restrict__ segmax,
                           float* __restrict__ segsum, int nE) {
  int i = blockIdx.x * blockDim.x + threadIdx.x;
  if (i >= nE * NHEAD) return;
  int e = i >> 2, hh = i & 3;
  int s = ei[e], d = ei[nE + e];
  float v = lrelu(asrc[s * NHEAD + hh] + adst[d * NHEAD + hh]);
  atomicAdd(&segsum[d * NHEAD + hh], __expf(v - segmax[d * NHEAD + hh]));
}

// out[n,c] = bias[c] + h[n,c] * att_self   (plain stores; runs before edge scatter)
__global__ void k_out_init(const float* __restrict__ h, const float* __restrict__ asrc,
                           const float* __restrict__ adst, const float* __restrict__ segmax,
                           const float* __restrict__ segsum, const float* __restrict__ bias,
                           float* __restrict__ out, int n) {
  int i = blockIdx.x * blockDim.x + threadIdx.x;
  if (i >= n * HD) return;
  int node = i >> 8, c = i & 255, hh = c >> 6;
  int ih = node * NHEAD + hh;
  float es = lrelu(asrc[ih] + adst[ih]);
  float att = __expf(es - segmax[ih]) / segsum[ih];
  float b = bias ? bias[c] : 0.f;
  out[i] = b + h[i] * att;
}

__global__ __launch_bounds__(256) void k_edge_scatter(
    const int* __restrict__ ei, const float* __restrict__ h, const float* __restrict__ asrc,
    const float* __restrict__ adst, const float* __restrict__ segmax,
    const float* __restrict__ segsum, float* __restrict__ out, int nE) {
  int c = threadIdx.x;  // 0..255 channel
  int hh = c >> 6;
  int e0 = blockIdx.x * 16;
#pragma unroll
  for (int j = 0; j < 16; ++j) {
    int e = e0 + j;
    if (e >= nE) return;
    int s = ei[e], d = ei[nE + e];
    int ih = d * NHEAD + hh;
    float v = lrelu(asrc[s * NHEAD + hh] + adst[ih]);
    float att = __expf(v - segmax[ih]) / segsum[ih];
    atomicAdd(&out[(size_t)d * HD + c], h[(size_t)s * HD + c] * att);
  }
}

// ---------------- BatchNorm ----------------

__global__ void k_bn_stats(const float* __restrict__ X, float* __restrict__ sum,
                           float* __restrict__ sumsq, int n, int C) {
  int c = threadIdx.x;  // blockDim.x == C
  int r0 = blockIdx.x * 512;
  int r1 = min(r0 + 512, n);
  float s = 0.f, q = 0.f;
  for (int r = r0; r < r1; ++r) {
    float v = X[(size_t)r * C + c];
    s += v;
    q += v * v;
  }
  atomicAdd(&sum[c], s);
  atomicAdd(&sumsq[c], q);
}

__global__ void k_bn_finalize(float* sum, float* sumsq, int n, int C) {
  int c = threadIdx.x;
  if (c >= C) return;
  float inv = 1.f / (float)n;
  float mu = sum[c] * inv;
  float var = sumsq[c] * inv - mu * mu;
  sum[c] = mu;
  sumsq[c] = __frsqrt_rn(var + BN_EPS);
}

// BN apply + leaky-relu, emit bf16 features for next GEMM
__global__ void k_bn_apply_bf16(const float* __restrict__ X, const float* __restrict__ mu,
                                const float* __restrict__ rs, const float* __restrict__ g,
                                const float* __restrict__ be, __bf16* __restrict__ Y,
                                int total, int C) {
  int i = blockIdx.x * blockDim.x + threadIdx.x;
  if (i >= total) return;
  int c = i & (C - 1);
  float v = (X[i] - mu[c]) * rs[c] * g[c] + be[c];
  Y[i] = (__bf16)lrelu(v);
}

__global__ void k_bn_apply_f32(float* __restrict__ X, const float* __restrict__ mu,
                               const float* __restrict__ rs, const float* __restrict__ g,
                               const float* __restrict__ be, int total, int C) {
  int i = blockIdx.x * blockDim.x + threadIdx.x;
  if (i >= total) return;
  int c = i & (C - 1);
  X[i] = (X[i] - mu[c]) * rs[c] * g[c] + be[c];
}

// head average + bias (layer 2, concat=False)
__global__ void k_headmean(const float* __restrict__ out, const float* __restrict__ b2,
                           float* __restrict__ hfin, int n) {
  int i = blockIdx.x * blockDim.x + threadIdx.x;
  if (i >= n * HID) return;
  int node = i >> 6, d = i & 63;
  const float* r = out + (size_t)node * HD + d;
  hfin[i] = 0.25f * (r[0] + r[64] + r[128] + r[192]) + b2[d];
}

// ---------------- fused edge MLP (WMMA) ----------------
// 4 waves/block; each wave owns 16 edges. ef(16x224 bf16) built in LDS, then
// 16x224@224x128 -> leaky -> 16x128@128x64 -> leaky -> 16x64@64x1, all per wave.
// K-outer / col-tile-inner loops keep all accumulators live and load each
// A fragment once per K-step (8x fewer ds_loads in layer 1).
#define MLPW 4
__global__ __launch_bounds__(128) void k_edge_mlp(
    const int* __restrict__ ei, const float* __restrict__ eattr,
    const float* __restrict__ hfin, const __bf16* __restrict__ Wt1,
    const float* __restrict__ bm1, const __bf16* __restrict__ Wt2,
    const float* __restrict__ bm2, const float* __restrict__ wm3,
    const float* __restrict__ bm3, float* __restrict__ out, int nE) {
  __shared__ alignas(16) __bf16 s_ef[MLPW][16][224];
  __shared__ alignas(16) __bf16 s_e2[MLPW][16][128];
  __shared__ alignas(16) __bf16 s_e3[MLPW][16][64];
  int w = threadIdx.x >> 5, lane = threadIdx.x & 31;
  int hh = lane >> 4, ln = lane & 15;
  int ebase = blockIdx.x * (MLPW * 16) + w * 16;

  // phase 1: gather + build edge features (bf16) into LDS
  for (int t = lane; t < 16 * 224; t += 32) {
    int i = t / 224, k = t - (t / 224) * 224;
    int e = min(ebase + i, nE - 1);
    int s = ei[e], d = ei[nE + e];
    float v;
    if (k < 64)
      v = hfin[(size_t)s * HID + k];
    else if (k < 128)
      v = hfin[(size_t)d * HID + (k - 64)];
    else if (k < 192) {
      int kk = k - 128;
      v = fabsf(hfin[(size_t)s * HID + kk] - hfin[(size_t)d * HID + kk]);
    } else if (k == 192)
      v = eattr[e];
    else
      v = 0.0f;
    s_ef[w][i][k] = (__bf16)v;
  }
  __syncthreads();

  // MLP layer 1: 16x224 @ 224x128 (8 live accumulators, A loaded once per kb)
  {
    v8f acc[8];
#pragma unroll
    for (int ct = 0; ct < 8; ++ct) acc[ct] = v8f{};
#pragma unroll
    for (int kb = 0; kb < 224; kb += 32) {
      v16bf a = ld_frag(&s_ef[w][ln][kb + 8 * hh]);
#pragma unroll
      for (int ct = 0; ct < 8; ++ct) {
        v16bf b = ld_frag(&Wt1[(size_t)(ct * 16 + ln) * 224 + kb + 8 * hh]);
        acc[ct] = __builtin_amdgcn_wmma_f32_16x16x32_bf16(false, a, false, b, (short)0,
                                                          acc[ct], false, false);
      }
    }
#pragma unroll
    for (int ct = 0; ct < 8; ++ct) {
      int col = ct * 16 + ln;
      float bias = bm1[col];
#pragma unroll
      for (int r = 0; r < 8; ++r) s_e2[w][8 * hh + r][col] = (__bf16)lrelu(acc[ct][r] + bias);
    }
  }
  __syncthreads();

  // MLP layer 2: 16x128 @ 128x64 (4 live accumulators)
  {
    v8f acc[4];
#pragma unroll
    for (int ct = 0; ct < 4; ++ct) acc[ct] = v8f{};
#pragma unroll
    for (int kb = 0; kb < 128; kb += 32) {
      v16bf a = ld_frag(&s_e2[w][ln][kb + 8 * hh]);
#pragma unroll
      for (int ct = 0; ct < 4; ++ct) {
        v16bf b = ld_frag(&Wt2[(size_t)(ct * 16 + ln) * 128 + kb + 8 * hh]);
        acc[ct] = __builtin_amdgcn_wmma_f32_16x16x32_bf16(false, a, false, b, (short)0,
                                                          acc[ct], false, false);
      }
    }
#pragma unroll
    for (int ct = 0; ct < 4; ++ct) {
      int col = ct * 16 + ln;
      float bias = bm2[col];
#pragma unroll
      for (int r = 0; r < 8; ++r) s_e3[w][8 * hh + r][col] = (__bf16)lrelu(acc[ct][r] + bias);
    }
  }
  __syncthreads();

  // MLP layer 3: 16x64 @ 64x1 (VALU dot, cross-half reduce via shfl)
  float part = 0.f;
#pragma unroll
  for (int k = 0; k < 32; ++k) {
    int kk = hh * 32 + k;
    part += (float)s_e3[w][ln][kk] * wm3[kk];
  }
  part += __shfl_xor(part, 16, 32);
  int eo = ebase + ln;
  if (hh == 0 && eo < nE) out[eo] = part + bm3[0];
}

// ---------------- launch ----------------

extern "C" void kernel_launch(void* const* d_in, const int* in_sizes, int n_in,
                              void* d_out, int out_size, void* d_ws, size_t ws_size,
                              hipStream_t stream) {
  const float* x = (const float*)d_in[0];
  const int* ei = (const int*)d_in[1];
  const float* eattr = (const float*)d_in[2];
  const float* W[3] = {(const float*)d_in[3], (const float*)d_in[7], (const float*)d_in[11]};
  const float* As[3] = {(const float*)d_in[4], (const float*)d_in[8], (const float*)d_in[12]};
  const float* Ad[3] = {(const float*)d_in[5], (const float*)d_in[9], (const float*)d_in[13]};
  const float* Bb[3] = {(const float*)d_in[6], (const float*)d_in[10], (const float*)d_in[14]};
  const float* G[3] = {(const float*)d_in[15], (const float*)d_in[17], (const float*)d_in[19]};
  const float* Be[3] = {(const float*)d_in[16], (const float*)d_in[18], (const float*)d_in[20]};
  const float* Wm1 = (const float*)d_in[21];
  const float* bm1 = (const float*)d_in[22];
  const float* Wm2 = (const float*)d_in[23];
  const float* bm2 = (const float*)d_in[24];
  const float* Wm3 = (const float*)d_in[25];
  const float* bm3 = (const float*)d_in[26];

  const int n = in_sizes[0] / 128;  // 50000 nodes
  const int nE = in_sizes[1] / 2;   // 800000 edges

  char* p = (char*)d_ws;
  auto alloc = [&](size_t bytes) {
    void* r = (void*)p;
    p += (bytes + 255) & ~(size_t)255;
    return r;
  };
  __bf16* hbf = (__bf16*)alloc((size_t)n * HD * 2);
  float* hlin = (float*)alloc((size_t)n * HD * 4);
  float* outb = (float*)alloc((size_t)n * HD * 4);
  float* asrc = (float*)alloc((size_t)n * NHEAD * 4);
  float* adst = (float*)alloc((size_t)n * NHEAD * 4);
  void* segmx = alloc((size_t)n * NHEAD * 4);
  float* segsm = (float*)alloc((size_t)n * NHEAD * 4);
  float* hfin = (float*)alloc((size_t)n * HID * 4);
  float* bnmu = (float*)alloc(HD * 4);
  float* bnrs = (float*)alloc(HD * 4);
  __bf16* Wt0 = (__bf16*)alloc((size_t)256 * 128 * 2);
  __bf16* Wt1 = (__bf16*)alloc((size_t)256 * 256 * 2);
  __bf16* Wt2 = (__bf16*)alloc((size_t)256 * 256 * 2);
  __bf16* Wtm1 = (__bf16*)alloc((size_t)128 * 224 * 2);
  __bf16* Wtm2 = (__bf16*)alloc((size_t)64 * 128 * 2);

  // pack weights (transposed, bf16, K zero-padded)
  k_pack_wt<<<(256 * 128 + 255) / 256, 256, 0, stream>>>(W[0], Wt0, 128, 256, 128);
  k_pack_wt<<<(256 * 256 + 255) / 256, 256, 0, stream>>>(W[1], Wt1, 256, 256, 256);
  k_pack_wt<<<(256 * 256 + 255) / 256, 256, 0, stream>>>(W[2], Wt2, 256, 256, 256);
  k_pack_wt<<<(128 * 224 + 255) / 256, 256, 0, stream>>>(Wm1, Wtm1, 193, 128, 224);
  k_pack_wt<<<(64 * 128 + 255) / 256, 256, 0, stream>>>(Wm2, Wtm2, 128, 64, 128);

  // layer 0 input: cast x to bf16
  k_cast_bf16<<<((n * 128) + 255) / 256, 256, 0, stream>>>(x, hbf, n * 128);

  int gNH = (n * NHEAD + 255) / 256;
  int gEH = (nE * NHEAD + 255) / 256;
  int gTiles = ((n / 16) * (HD / 64) + 7) / 8;  // 16x64 strips, 8 waves/block

  for (int l = 0; l < 3; ++l) {
    // 1) linear transform via WMMA
    if (l == 0)
      k_gemm_bf16<128><<<gTiles, 256, 0, stream>>>(hbf, Wt0, hlin, n, HD);
    else
      k_gemm_bf16<256><<<gTiles, 256, 0, stream>>>(hbf, (l == 1) ? Wt1 : Wt2, hlin, n, HD);
    // 2) attention coefficients + segment softmax (self-loops included)
    k_alpha<<<gNH, 256, 0, stream>>>(hlin, As[l], Ad[l], asrc, adst, (unsigned*)segmx, n);
    k_edge_max<<<gEH, 256, 0, stream>>>(ei, asrc, adst, (unsigned*)segmx, nE);
    k_node_expinit<<<gNH, 256, 0, stream>>>(asrc, adst, (unsigned*)segmx, segsm, n);
    k_edge_sum<<<gEH, 256, 0, stream>>>(ei, asrc, adst, (const float*)segmx, segsm, nE);
    // 3) weighted aggregation
    k_out_init<<<(n * HD + 255) / 256, 256, 0, stream>>>(
        hlin, asrc, adst, (const float*)segmx, segsm, (l < 2) ? Bb[l] : nullptr, outb, n);
    k_edge_scatter<<<(nE + 15) / 16, 256, 0, stream>>>(ei, hlin, asrc, adst,
                                                       (const float*)segmx, segsm, outb, nE);
    // 4) batchnorm (+ leaky & bf16 cast for next layer)
    if (l < 2) {
      k_zero<<<1, 256, 0, stream>>>(bnmu, HD);
      k_zero<<<1, 256, 0, stream>>>(bnrs, HD);
      k_bn_stats<<<(n + 511) / 512, HD, 0, stream>>>(outb, bnmu, bnrs, n, HD);
      k_bn_finalize<<<1, HD, 0, stream>>>(bnmu, bnrs, n, HD);
      k_bn_apply_bf16<<<(n * HD + 255) / 256, 256, 0, stream>>>(outb, bnmu, bnrs, G[l],
                                                                Be[l], hbf, n * HD, HD);
    } else {
      k_headmean<<<(n * HID + 255) / 256, 256, 0, stream>>>(outb, Bb[2], hfin, n);
      k_zero<<<1, 256, 0, stream>>>(bnmu, HID);
      k_zero<<<1, 256, 0, stream>>>(bnrs, HID);
      k_bn_stats<<<(n + 511) / 512, HID, 0, stream>>>(hfin, bnmu, bnrs, n, HID);
      k_bn_finalize<<<1, HID, 0, stream>>>(bnmu, bnrs, n, HID);
      k_bn_apply_f32<<<(n * HID + 255) / 256, 256, 0, stream>>>(hfin, bnmu, bnrs, G[2],
                                                                Be[2], n * HID, HID);
    }
  }

  // fused edge MLP -> logits
  k_edge_mlp<<<(nE + MLPW * 16 - 1) / (MLPW * 16), MLPW * 32, 0, stream>>>(
      ei, eattr, hfin, Wtm1, bm1, Wtm2, bm2, Wm3, bm3, (float*)d_out, nE);
}
